// InterAttention_87686052315706
// MI455X (gfx1250) — matrix operand order repaired and checked by reading, not compiled, not run on hardware
//
#include <hip/hip_runtime.h>

typedef unsigned short u16;
typedef __attribute__((ext_vector_type(16))) __bf16 v16bf;
typedef __attribute__((ext_vector_type(8)))  float  v8f;

union AFrag { v16bf v; uint4 u[2]; };
union AccU  { v8f v; float f[8]; };

__device__ __forceinline__ u16 f2bf(float x) {
  unsigned u = __float_as_uint(x);
  u += 0x7fffu + ((u >> 16) & 1u);   // round-to-nearest-even on raw bits
  return (u16)(u >> 16);
}

// One 16-byte async DMA: LDS[lds_off] = global[g]   (tracked by ASYNCcnt)
__device__ __forceinline__ void async_b128(unsigned lds_off, const void* g) {
  asm volatile("global_load_async_to_lds_b128 %0, %1, off"
               :: "v"(lds_off), "v"(g) : "memory");
}
__device__ __forceinline__ void wait_async0() {
  asm volatile("s_wait_asynccnt 0x0" ::: "memory");
}
__device__ __forceinline__ unsigned lds_off(const void* p) {
  return (unsigned)(unsigned long long)p;   // low 32 bits = LDS offset
}

// ---------------------------------------------------------------------------
// Generic bf16 WMMA GEMM:  C[M,N] = act(A[M,K] @ B[K,N] + bias) * mask
//   A  : bf16 row-major [M, lda]      (per-batch stride aBatch elements)
//   Bt : bf16 row-major B^T [N, ldb]  (per-batch stride bBatch elements)
// Epilogues (all optional): fp32 row-major, bf16 row-major (stride ldoH),
// bf16 transposed per-"trRows"-batch ([N, trRows] tiles).
// Block tile 128x128, 8 waves of 64x32, K step 32.
// Double-buffered LDS filled by global_load_async_to_lds_b128 (ASYNCcnt).
// ---------------------------------------------------------------------------
__global__ __launch_bounds__(256) void gemm_bf16_wmma(
    const u16* __restrict__ A,  int lda, long long aBatch,
    const u16* __restrict__ Bt, int ldb, long long bBatch,
    int K,
    const float* __restrict__ bias,
    const int*  __restrict__ mask,     // per-row token mask (flat GEMMs only)
    int doLrelu,
    float* __restrict__ outF, int ldoF, long long oFBatch,
    u16*  __restrict__ outH, int ldoH, long long oHBatch,
    u16*  __restrict__ outT, int ldT, int trRows, long long trBatch)
{
  __shared__ __align__(16) u16 sA[2][128 * 32];
  __shared__ __align__(16) u16 sB[2][128 * 32];

  const int t     = threadIdx.x;
  const int lane  = t & 31;
  const int w     = t >> 5;
  const int waveM = w & 1;      // 2 waves cover 128 rows
  const int waveN = w >> 1;     // 4 waves cover 128 cols
  const int half  = lane >> 4;
  const int r     = lane & 15;

  const int m0 = blockIdx.y << 7;
  const int n0 = blockIdx.x << 7;
  const int bz = blockIdx.z;

  const int srow = t >> 1;          // 0..127
  const int scol = (t & 1) << 4;    // 0 or 16 (elements)

  // per-thread global staging pointers (advance along K)
  const u16* gA = A  + (long long)bz * aBatch + (size_t)(m0 + srow) * lda + scol;
  const u16* gB = Bt + (long long)bz * bBatch + (size_t)(n0 + srow) * ldb + scol;

  v8f acc[4][2];
#pragma unroll
  for (int i = 0; i < 4; ++i)
#pragma unroll
    for (int j = 0; j < 2; ++j) acc[i][j] = {};

  const int nk = K >> 5;

  // preload stage 0 into buffer 0
  {
    async_b128(lds_off(&sA[0][srow * 32 + scol]),     gA);
    async_b128(lds_off(&sA[0][srow * 32 + scol + 8]), gA + 8);
    async_b128(lds_off(&sB[0][srow * 32 + scol]),     gB);
    async_b128(lds_off(&sB[0][srow * 32 + scol + 8]), gB + 8);
  }

  for (int i = 0; i < nk; ++i) {
    wait_async0();        // my stage-i DMA writes have landed in LDS
    __syncthreads();      // everyone's have; everyone done reading buf (i+1)&1

    if (i + 1 < nk) {     // prefetch stage i+1 while computing stage i
      const int nb = (i + 1) & 1;
      const int k1 = (i + 1) << 5;
      async_b128(lds_off(&sA[nb][srow * 32 + scol]),     gA + k1);
      async_b128(lds_off(&sA[nb][srow * 32 + scol + 8]), gA + k1 + 8);
      async_b128(lds_off(&sB[nb][srow * 32 + scol]),     gB + k1);
      async_b128(lds_off(&sB[nb][srow * 32 + scol + 8]), gB + k1 + 8);
    }

    const u16* cA = sA[i & 1];
    const u16* cB = sB[i & 1];

    AFrag fa[4], fb[2];
#pragma unroll
    for (int mi = 0; mi < 4; ++mi) {               // A 16x32: lane=row r, two 8-K chunks
      int ar = waveM * 64 + mi * 16 + r;
      fa[mi].u[0] = *(const uint4*)&cA[ar * 32 + half * 8];
      fa[mi].u[1] = *(const uint4*)&cA[ar * 32 + 16 + half * 8];
    }
#pragma unroll
    for (int ni = 0; ni < 2; ++ni) {               // B 32x16: lane=col r, two 8-K chunks
      int bc = waveN * 32 + ni * 16 + r;
      fb[ni].u[0] = *(const uint4*)&cB[bc * 32 + half * 8];
      fb[ni].u[1] = *(const uint4*)&cB[bc * 32 + 16 + half * 8];
    }
#pragma unroll
    for (int mi = 0; mi < 4; ++mi)
#pragma unroll
      for (int ni = 0; ni < 2; ++ni)
        acc[mi][ni] = __builtin_amdgcn_wmma_f32_16x16x32_bf16(
            false, fa[mi].v, false, fb[ni].v, (short)0, acc[mi][ni], false, false);
  }

  float* outFb = outF ? outF + (long long)bz * oFBatch : nullptr;
  u16*  outHb = outH ? outH + (long long)bz * oHBatch : nullptr;

#pragma unroll
  for (int mi = 0; mi < 4; ++mi) {
    const int mbase = m0 + waveM * 64 + mi * 16 + half * 8;  // C: m = vgpr + 8*half
#pragma unroll
    for (int ni = 0; ni < 2; ++ni) {
      const int n = n0 + waveN * 32 + ni * 16 + r;           // C: n = lane%16
      AccU a; a.v = acc[mi][ni];
      const float bv = bias ? bias[n] : 0.0f;
      float xv[8];
#pragma unroll
      for (int v = 0; v < 8; ++v) {
        float x = a.f[v] + bv;
        if (doLrelu) x = (x >= 0.0f) ? x : 0.01f * x;
        if (mask)    x *= (float)mask[mbase + v];
        xv[v] = x;
      }
      if (outFb) {
#pragma unroll
        for (int v = 0; v < 8; ++v)
          outFb[(size_t)(mbase + v) * ldoF + n] = xv[v];
      }
      if (outHb) {
#pragma unroll
        for (int v = 0; v < 8; ++v)
          outHb[(size_t)(mbase + v) * ldoH + n] = f2bf(xv[v]);
      }
      if (outT) {   // transposed store: 8 consecutive m, fixed n -> one b128
        int bb = mbase / trRows;
        int ml = mbase - bb * trRows;
        uint4 pk;
        pk.x = (unsigned)f2bf(xv[0]) | ((unsigned)f2bf(xv[1]) << 16);
        pk.y = (unsigned)f2bf(xv[2]) | ((unsigned)f2bf(xv[3]) << 16);
        pk.z = (unsigned)f2bf(xv[4]) | ((unsigned)f2bf(xv[5]) << 16);
        pk.w = (unsigned)f2bf(xv[6]) | ((unsigned)f2bf(xv[7]) << 16);
        *(uint4*)&outT[(size_t)bb * trBatch + (size_t)n * ldT + ml] = pk;
      }
    }
  }
}

// Row softmax over L=512 with pairwise validity mask; bf16 probabilities out.
__global__ __launch_bounds__(256) void softmax_mask_rows(
    const float* __restrict__ S, const int* __restrict__ mRow,
    const int* __restrict__ mCol, u16* __restrict__ out)
{
  const int gid = blockIdx.x;       // 0 .. B*L-1
  const int b   = gid >> 9;
  const int t   = threadIdx.x;
  const float* srow = S + (size_t)gid * 512;
  const int mr = mRow[gid];
  const int c0 = t, c1 = t + 256;
  const int ok0 = mr && mCol[(b << 9) + c0];
  const int ok1 = mr && mCol[(b << 9) + c1];
  float x0 = ok0 ? srow[c0] : -1e10f;
  float x1 = ok1 ? srow[c1] : -1e10f;

  __shared__ float red[256];
  red[t] = fmaxf(x0, x1);
  __syncthreads();
  for (int s = 128; s > 0; s >>= 1) { if (t < s) red[t] = fmaxf(red[t], red[t + s]); __syncthreads(); }
  float mx = red[0];
  __syncthreads();
  float e0 = __expf(x0 - mx), e1 = __expf(x1 - mx);
  red[t] = e0 + e1;
  __syncthreads();
  for (int s = 128; s > 0; s >>= 1) { if (t < s) red[t] += red[t + s]; __syncthreads(); }
  float inv = 1.0f / red[0];
  u16* orow = out + (size_t)gid * 512;
  orow[c0] = f2bf(ok0 ? e0 * inv : 0.0f);
  orow[c1] = f2bf(ok1 ? e1 * inv : 0.0f);
}

__global__ __launch_bounds__(256) void cvt_bf16(
    const float* __restrict__ in, u16* __restrict__ out, int n4)
{
  int i = blockIdx.x * 256 + threadIdx.x;
  if (i >= n4) return;
  float4 v = ((const float4*)in)[i];
  uint2 o;
  o.x = (unsigned)f2bf(v.x) | ((unsigned)f2bf(v.y) << 16);
  o.y = (unsigned)f2bf(v.z) | ((unsigned)f2bf(v.w) << 16);
  ((uint2*)out)[i] = o;
}

// Wt[n*K + k] = bf16(W[k*N + n])  (weights to B^T layout)
__global__ __launch_bounds__(256) void cvt_transpose(
    const float* __restrict__ W, u16* __restrict__ Wt, int K, int N)
{
  int i = blockIdx.x * 256 + threadIdx.x;
  if (i >= K * N) return;
  int k = i / N, n = i - k * N;
  Wt[(size_t)n * K + k] = f2bf(W[i]);
}

// Copy rm (bf16 [row,512]) into left half of xcat ([row,1024]); 8 elems/thread.
__global__ __launch_bounds__(256) void copy_to_xcat(
    const u16* __restrict__ src, u16* __restrict__ dst)
{
  size_t i = (size_t)blockIdx.x * 256 + threadIdx.x;  // one per 8-elem chunk
  size_t row = i >> 6;
  size_t c   = (i & 63) << 3;
  *(uint4*)&dst[row * 1024 + c] = *(const uint4*)&src[row * 512 + c];
}

static void launch_gemm(hipStream_t s,
                        const u16* A, int lda, long long aB,
                        const u16* Bt, int ldb, long long bB,
                        int M, int N, int K, int nb,
                        const float* bias, const int* mask, int lrelu,
                        float* oF, int ldoF, long long oFB,
                        u16* oH, int ldoH, long long oHB,
                        u16* oT, int ldT, int trRows, long long trB)
{
  dim3 g(N / 128, M / 128, nb), b(256);
  gemm_bf16_wmma<<<g, b, 0, s>>>(A, lda, aB, Bt, ldb, bB, K, bias, mask, lrelu,
                                 oF, ldoF, oFB, oH, ldoH, oHB, oT, ldT, trRows, trB);
}

extern "C" void kernel_launch(void* const* d_in, const int* in_sizes, int n_in,
                              void* d_out, int out_size, void* d_ws, size_t ws_size,
                              hipStream_t stream)
{
  (void)in_sizes; (void)n_in; (void)out_size; (void)ws_size;
  const float* r1    = (const float*)d_in[0];
  const float* r2    = (const float*)d_in[1];
  const int*   mask1 = (const int*)d_in[2];
  const int*   mask2 = (const int*)d_in[3];
  const float* W1  = (const float*)d_in[4];
  const float* b1  = (const float*)d_in[5];
  const float* W2  = (const float*)d_in[6];
  const float* b2  = (const float*)d_in[7];
  const float* Wc1 = (const float*)d_in[8];
  const float* bc1 = (const float*)d_in[9];
  const float* Wc2 = (const float*)d_in[10];
  const float* bc2 = (const float*)d_in[11];

  constexpr int B = 64, L = 512, D = 512;
  constexpr int BL = B * L;                         // 32768
  constexpr long long BLD = (long long)BL * D;      // 16,777,216
  constexpr long long LL  = (long long)L * L;       // 262,144 (per-batch)

  char* base = (char*)d_ws; size_t off = 0;
  auto alloc = [&](size_t bytes) -> void* {
    void* q = base + off;
    off = (off + bytes + 255) & ~(size_t)255;
    return q;
  };

  u16*  r1bf = (u16*)alloc((size_t)BLD * 2);
  u16*  r2bf = (u16*)alloc((size_t)BLD * 2);
  u16*  W1t  = (u16*)alloc((size_t)512 * 512 * 2);
  u16*  W2t  = (u16*)alloc((size_t)512 * 512 * 2);
  u16*  Wc1t = (u16*)alloc((size_t)1024 * 512 * 2);
  u16*  Wc2t = (u16*)alloc((size_t)512 * 512 * 2);
  u16*  h    = (u16*)alloc((size_t)BLD * 2);
  u16*  r1m  = (u16*)alloc((size_t)BLD * 2);
  u16*  r2m  = (u16*)alloc((size_t)BLD * 2);
  u16*  r1mT = (u16*)alloc((size_t)BLD * 2);
  u16*  r2mT = (u16*)alloc((size_t)BLD * 2);
  float* S   = (float*)alloc((size_t)B * LL * 4);   // 64 MB, later reused as xcat
  u16*  o1   = r1bf;                                // reuse: r1bf dead after MLP
  u16*  o2T  = r2bf;
  u16*  xcat = (u16*)S;                             // [32768, 1024] bf16

  // ---- convert activations & weights to bf16 (weights -> B^T layout) ----
  {
    int n4 = (int)(BLD / 4);
    cvt_bf16<<<n4 / 256, 256, 0, stream>>>(r1, r1bf, n4);
    cvt_bf16<<<n4 / 256, 256, 0, stream>>>(r2, r2bf, n4);
  }
  cvt_transpose<<<(512 * 512) / 256, 256, 0, stream>>>(W1,  W1t,  512,  512);
  cvt_transpose<<<(512 * 512) / 256, 256, 0, stream>>>(W2,  W2t,  512,  512);
  cvt_transpose<<<(1024 * 512) / 256, 256, 0, stream>>>(Wc1, Wc1t, 1024, 512);
  cvt_transpose<<<(512 * 512) / 256, 256, 0, stream>>>(Wc2, Wc2t, 512,  512);

  // ---- shared MLP: rm = lrelu(lrelu(r@W1+b1)@W2+b2) * mask ----
  launch_gemm(stream, r1bf, 512, 0, W1t, 512, 0, BL, 512, 512, 1,
              b1, nullptr, 1, nullptr, 0, 0, h, 512, 0, nullptr, 0, 0, 0);
  launch_gemm(stream, h, 512, 0, W2t, 512, 0, BL, 512, 512, 1,
              b2, mask1, 1, nullptr, 0, 0, r1m, 512, 0, r1mT, 512, 512, LL);
  launch_gemm(stream, r2bf, 512, 0, W1t, 512, 0, BL, 512, 512, 1,
              b1, nullptr, 1, nullptr, 0, 0, h, 512, 0, nullptr, 0, 0, 0);
  launch_gemm(stream, h, 512, 0, W2t, 512, 0, BL, 512, 512, 1,
              b2, mask2, 1, nullptr, 0, 0, r2m, 512, 0, r2mT, 512, 512, LL);

  // ---- scores + masked softmax (axis-2 on S, axis-1 via S^T) ----
  launch_gemm(stream, r1m, 512, LL, r2m, 512, LL, 512, 512, 512, B,
              nullptr, nullptr, 0, S, 512, LL, nullptr, 0, 0, nullptr, 0, 0, 0);
  softmax_mask_rows<<<BL, 256, 0, stream>>>(S, mask1, mask2, o1);
  launch_gemm(stream, r2m, 512, LL, r1m, 512, LL, 512, 512, 512, B,
              nullptr, nullptr, 0, S, 512, LL, nullptr, 0, 0, nullptr, 0, 0, 0);
  softmax_mask_rows<<<BL, 256, 0, stream>>>(S, mask2, mask1, o2T);

  // ---- r1 side: xcat = [r1m | o1 @ r2m], then compare MLP -> d_out[0:] ----
  copy_to_xcat<<<(BL * 64) / 256, 256, 0, stream>>>(r1m, xcat);
  launch_gemm(stream, o1, 512, LL, r2mT, 512, LL, 512, 512, 512, B,
              nullptr, nullptr, 0, nullptr, 0, 0, xcat + 512, 1024, (long long)512 * 1024,
              nullptr, 0, 0, 0);
  launch_gemm(stream, xcat, 1024, 0, Wc1t, 1024, 0, BL, 512, 1024, 1,
              bc1, nullptr, 1, nullptr, 0, 0, h, 512, 0, nullptr, 0, 0, 0);
  launch_gemm(stream, h, 512, 0, Wc2t, 512, 0, BL, 512, 512, 1,
              bc2, mask1, 1, (float*)d_out, 512, 0, nullptr, 0, 0, nullptr, 0, 0, 0);

  // ---- r2 side: xcat = [r2m | o2T @ r1m], compare MLP -> d_out[BLD:] ----
  copy_to_xcat<<<(BL * 64) / 256, 256, 0, stream>>>(r2m, xcat);
  launch_gemm(stream, o2T, 512, LL, r1mT, 512, LL, 512, 512, 512, B,
              nullptr, nullptr, 0, nullptr, 0, 0, xcat + 512, 1024, (long long)512 * 1024,
              nullptr, 0, 0, 0);
  launch_gemm(stream, xcat, 1024, 0, Wc1t, 1024, 0, BL, 512, 1024, 1,
              bc1, nullptr, 1, nullptr, 0, 0, h, 512, 0, nullptr, 0, 0, 0);
  launch_gemm(stream, h, 512, 0, Wc2t, 512, 0, BL, 512, 512, 1,
              bc2, mask2, 1, (float*)d_out + BLD, 512, 0, nullptr, 0, 0, nullptr, 0, 0, 0);
}
